// Transformer_24378234372553
// MI455X (gfx1250) — compile-verified, matrix operand order checked by reading, not run on hardware
//
#include <hip/hip_runtime.h>

typedef __bf16 bf16;
typedef __attribute__((ext_vector_type(16))) __bf16 v16bf;
typedef __attribute__((ext_vector_type(8)))  __bf16 v8bf;
typedef __attribute__((ext_vector_type(8)))  float  v8f;
typedef __attribute__((ext_vector_type(4)))  unsigned int u32x4;
typedef __attribute__((ext_vector_type(8)))  int i32x8;
typedef __attribute__((ext_vector_type(4)))  int i32x4;

#define BB 2
#define SS 2048
#define EE 512
#define HH 8
#define HDD 64
#define FFN 2048
#define LLY 2
#define VV 32000
#define T3 1536    // 3*E
#define KSTEP 32
#define ROWB 80    // LDS panel row stride: 64B data + 16B TDM pad (bank rotate, 16B aligned)

// intra-wave LDS ordering fence (DS pipe is in-order per wave; stop compiler reordering)
#define LDS_FENCE() asm volatile("s_wait_dscnt 0x0" ::: "memory")

// ---------------------------------------------------------------------------
// TDM 2D panel load: rows x 32 bf16 elements, padded 16B per 64B row in LDS.
// ---------------------------------------------------------------------------
#if __has_builtin(__builtin_amdgcn_tensor_load_to_lds)
#define HAVE_TDM 1
__device__ __forceinline__ void tdm_load_2d(const void* gptr, unsigned ldsOff,
                                            int dim0, int stride, int rows_total,
                                            int tile_rows) {
  unsigned long long ga = (unsigned long long)(uintptr_t)gptr;
  u32x4 g0;
  g0[0] = 1u;                                   // count=1 user descriptor
  g0[1] = ldsOff;                               // LDS byte address
  g0[2] = (unsigned)(ga & 0xffffffffull);       // global addr lo
  g0[3] = (unsigned)(ga >> 32) | (2u << 30);    // global addr hi + type=2
  i32x8 g1;
  // data_size=2B | pad_enable | pad_interval=16dw | pad_amount=4dw
  g1[0] = (int)((1u << 16) | (1u << 20) | (3u << 22) | (3u << 25));
  g1[1] = (int)((unsigned)(dim0 & 0xffff) << 16);                       // tensor_dim0 lo16
  g1[2] = (int)(((unsigned)dim0 >> 16) | ((unsigned)(rows_total & 0xffff) << 16));
  g1[3] = (int)(((unsigned)rows_total >> 16) | (32u << 16));            // dim1 hi | tile_dim0=32
  g1[4] = tile_rows;                                                    // tile_dim1 (tile_dim2=0)
  g1[5] = stride;                                                       // tensor_dim0_stride lo32
  g1[6] = 0;
  g1[7] = 0;
  i32x4 z4 = {0, 0, 0, 0};
#if defined(__clang_major__) && (__clang_major__ >= 23)
  i32x8 z8 = {0, 0, 0, 0, 0, 0, 0, 0};
  __builtin_amdgcn_tensor_load_to_lds(g0, g1, z4, z4, z8, 0);   // 6-arg form
#else
  __builtin_amdgcn_tensor_load_to_lds(g0, g1, z4, z4, 0);       // 5-arg form (ROCm 7.2)
#endif
}
#endif

// ---------------------------------------------------------------------------
// Position-reset scan + segment ids (tiny: 2 rows of 2048)
// ---------------------------------------------------------------------------
__global__ void k_scan(const int* __restrict__ tok, const int* __restrict__ start_id,
                       int* __restrict__ relpos, int* __restrict__ seg) {
  if (threadIdx.x != 0) return;
  int b = blockIdx.x;
  int sid = *start_id;
  int last = 0, c = 0;
  for (int s = 0; s < SS; ++s) {
    int t = tok[b * SS + s];
    int is0 = (t == sid) ? 1 : 0;
    c += is0;
    if (is0 || s == 0) last = s;
    relpos[b * SS + s] = s - last;
    seg[b * SS + s] = c;
  }
}

__global__ __launch_bounds__(256) void k_embed(const int* __restrict__ tok,
                                               const int* __restrict__ relpos,
                                               const float* __restrict__ we,
                                               const float* __restrict__ pe,
                                               float* __restrict__ x,
                                               bf16* __restrict__ xbf) {
  int row = blockIdx.x;
  int tk = tok[row];
  int rp = relpos[row];
  for (int e = threadIdx.x; e < EE; e += 256) {
    float v = we[(size_t)tk * EE + e] + pe[(size_t)rp * EE + e];
    x[(size_t)row * EE + e] = v;
    xbf[(size_t)row * EE + e] = (bf16)v;
  }
}

__global__ void k_transpose_bf(const float* __restrict__ in, bf16* __restrict__ out,
                               int K, int N) {
  size_t i = (size_t)blockIdx.x * blockDim.x + threadIdx.x;
  if (i >= (size_t)K * N) return;
  int n = (int)(i / K);
  int k = (int)(i % K);
  out[i] = (bf16)in[(size_t)k * N + n];
}

__global__ void k_f2bf(const float* __restrict__ in, bf16* __restrict__ out, size_t n) {
  size_t i = (size_t)blockIdx.x * blockDim.x + threadIdx.x;
  if (i < n) out[i] = (bf16)in[i];
}

// Pack V^T: vT[((b*H+h)*HD + d) * S + s] = kqv[(b*S+s)*T3 + h*192 + 128 + d]
__global__ __launch_bounds__(256) void k_packVT(const bf16* __restrict__ kqv,
                                                bf16* __restrict__ vT) {
  size_t i = (size_t)blockIdx.x * 256 + threadIdx.x;
  if (i >= (size_t)BB * HH * HDD * SS) return;
  int s = (int)(i & (SS - 1));
  int d = (int)((i >> 11) & (HDD - 1));
  int h = (int)((i >> 17) & (HH - 1));
  int b = (int)(i >> 20);
  vT[i] = kqv[((size_t)(b * SS + s)) * T3 + h * 192 + 128 + d];
}

// ---------------------------------------------------------------------------
// WMMA GEMM: C[M][N] = A[M][K] * Bt[N][K]^T (+bias)(+relu), bf16 in fp32 acc.
// Block: 8 waves = 128M x 256N tile; each wave 64x64 (4x4 accumulators).
// K panels (A:128x32, B:256x32) staged in LDS by the Tensor Data Mover,
// double buffered; fragments read from LDS as 16B ds loads.
// ---------------------------------------------------------------------------
__global__ __launch_bounds__(256) void k_gemm_bt(const bf16* __restrict__ A,
                                                 const bf16* __restrict__ Bt,
                                                 const float* __restrict__ bias,
                                                 float* __restrict__ outF,
                                                 bf16* __restrict__ outB,
                                                 int M, int N, int K, int relu) {
  __shared__ __align__(16) char sA[2][128 * ROWB];
  __shared__ __align__(16) char sB[2][256 * ROWB];
  int lane = threadIdx.x & 31;
  int wave = threadIdx.x >> 5;
  int idx = lane & 15, hi = lane >> 4;
  int wr = wave >> 2, wc = wave & 3;   // 2 x 4 waves
  int mbase = blockIdx.y * 128;
  int nbase = blockIdx.x * 256;

  v8f c[4][4];
#pragma unroll
  for (int i = 0; i < 4; ++i)
#pragma unroll
    for (int j = 0; j < 4; ++j) c[i][j] = (v8f){};

  int steps = K / KSTEP;
#ifdef HAVE_TDM
  if (wave == 0) {
    tdm_load_2d(A + (size_t)mbase * K, (unsigned)(uintptr_t)&sA[0][0],
                K, K, M - mbase, 128);
    tdm_load_2d(Bt + (size_t)nbase * K, (unsigned)(uintptr_t)&sB[0][0],
                K, K, N - nbase, 256);
  }
#endif
  for (int s = 0; s < steps; ++s) {
    int buf = s & 1;
#ifdef HAVE_TDM
    if (wave == 0) {
      if (s + 1 < steps) {
        int k0n = (s + 1) * KSTEP;
        tdm_load_2d(A + (size_t)mbase * K + k0n, (unsigned)(uintptr_t)&sA[buf ^ 1][0],
                    K - k0n, K, M - mbase, 128);
        tdm_load_2d(Bt + (size_t)nbase * K + k0n, (unsigned)(uintptr_t)&sB[buf ^ 1][0],
                    K - k0n, K, N - nbase, 256);
        __builtin_amdgcn_s_wait_tensorcnt(2);
      } else {
        __builtin_amdgcn_s_wait_tensorcnt(0);
      }
    }
#else
    {  // cooperative copy into the same padded LDS layout
      int k0 = s * KSTEP;
      int t = threadIdx.x;
      {
        int row = t >> 1, half = t & 1;
        const bf16* src = A + (size_t)(mbase + row) * K + k0 + half * 16;
        v8bf x0 = *(const v8bf*)(src);
        v8bf x1 = *(const v8bf*)(src + 8);
        char* dst = &sA[buf][row * ROWB + half * 32];
        *(v8bf*)dst = x0;
        *(v8bf*)(dst + 16) = x1;
      }
      {
        const bf16* src = Bt + (size_t)(nbase + t) * K + k0;
        char* dst = &sB[buf][t * ROWB];
#pragma unroll
        for (int j = 0; j < 4; ++j) *(v8bf*)(dst + j * 16) = *(const v8bf*)(src + j * 8);
      }
    }
#endif
    __syncthreads();

    v16bf af[4], bfr[4];
#pragma unroll
    for (int mt = 0; mt < 4; ++mt) {
      const char* base = &sA[buf][(wr * 64 + mt * 16 + idx) * ROWB];
      v8bf lo = *(const v8bf*)(base + hi * 16);
      v8bf hs = *(const v8bf*)(base + 32 + hi * 16);
#pragma unroll
      for (int i = 0; i < 8; ++i) { af[mt][i] = lo[i]; af[mt][8 + i] = hs[i]; }
    }
#pragma unroll
    for (int nt = 0; nt < 4; ++nt) {
      const char* base = &sB[buf][(wc * 64 + nt * 16 + idx) * ROWB];
      v8bf lo = *(const v8bf*)(base + hi * 16);
      v8bf hs = *(const v8bf*)(base + 32 + hi * 16);
#pragma unroll
      for (int i = 0; i < 8; ++i) { bfr[nt][i] = lo[i]; bfr[nt][8 + i] = hs[i]; }
    }
#pragma unroll
    for (int mt = 0; mt < 4; ++mt)
#pragma unroll
      for (int nt = 0; nt < 4; ++nt)
        c[mt][nt] = __builtin_amdgcn_wmma_f32_16x16x32_bf16(
            false, af[mt], false, bfr[nt], (short)0, c[mt][nt], false, false);
    __syncthreads();
  }

#pragma unroll
  for (int nt = 0; nt < 4; ++nt) {
    int n = nbase + wc * 64 + nt * 16 + idx;
    float bia = bias ? bias[n] : 0.0f;
#pragma unroll
    for (int mt = 0; mt < 4; ++mt) {
#pragma unroll
      for (int r = 0; r < 8; ++r) {
        int m = mbase + wr * 64 + mt * 16 + r + 8 * hi;
        float v = c[mt][nt][r] + bia;
        if (relu) v = v > 0.0f ? v : 0.0f;
        size_t o = (size_t)m * N + n;
        if (outF) outF[o] = v;
        if (outB) outB[o] = (bf16)v;
      }
    }
  }
}

// ---------------------------------------------------------------------------
// Flash attention: 4 waves/block, one (b,h,16-query tile) per wave.
// No block barriers (waves have different trip counts) — the P re-layout
// through LDS is intra-wave only (in-order DS pipe + explicit fence).
// ---------------------------------------------------------------------------
__global__ __launch_bounds__(128) void k_attn(const bf16* __restrict__ kqv,
                                              const bf16* __restrict__ vT,
                                              const int* __restrict__ seg,
                                              bf16* __restrict__ hoB) {
  __shared__ bf16 P[4][16][32];
  int lane = threadIdx.x & 31;
  int wave = threadIdx.x >> 5;
  int idx = lane & 15, hi = lane >> 4;
  int bid = blockIdx.x;
  int qg = bid & 31;            // (S/16)/4 = 32 groups
  int h = (bid >> 5) & 7;
  int b = bid >> 8;
  int qt = qg * 4 + wave;
  int qbase = qt * 16;
  const float scale = 0.125f;   // 1/sqrt(64)

  const bf16* qp = kqv + ((size_t)(b * SS + qbase + idx) * T3) + h * 192 + 64;
  v16bf qa0, qa1;
  {
    v8bf l0 = *(const v8bf*)(qp + hi * 8);
    v8bf h0 = *(const v8bf*)(qp + 16 + hi * 8);
    v8bf l1 = *(const v8bf*)(qp + 32 + hi * 8);
    v8bf h1 = *(const v8bf*)(qp + 48 + hi * 8);
#pragma unroll
    for (int i = 0; i < 8; ++i) {
      qa0[i] = l0[i]; qa0[8 + i] = h0[i];
      qa1[i] = l1[i]; qa1[8 + i] = h1[i];
    }
  }

  int segq[8];
#pragma unroll
  for (int r = 0; r < 8; ++r) segq[r] = seg[b * SS + qbase + r + 8 * hi];

  float mold[8], lsum[8];
  v8f acc[4];
#pragma unroll
  for (int r = 0; r < 8; ++r) { mold[r] = -1e30f; lsum[r] = 0.0f; }
#pragma unroll
  for (int t = 0; t < 4; ++t) acc[t] = (v8f){};

  const bf16* vbase = vT + ((size_t)(b * HH + h) * HDD) * SS;

  int npair = qt / 2 + 1;
  for (int p = 0; p < npair; ++p) {
    float s0[8], s1[8];
#pragma unroll
    for (int sub = 0; sub < 2; ++sub) {
      int kt = 2 * p + sub;
      const bf16* kp = kqv + ((size_t)(b * SS + kt * 16 + idx) * T3) + h * 192;
      v16bf kb0, kb1;
      {
        v8bf l0 = *(const v8bf*)(kp + hi * 8);
        v8bf h0 = *(const v8bf*)(kp + 16 + hi * 8);
        v8bf l1 = *(const v8bf*)(kp + 32 + hi * 8);
        v8bf h1 = *(const v8bf*)(kp + 48 + hi * 8);
#pragma unroll
        for (int i = 0; i < 8; ++i) {
          kb0[i] = l0[i]; kb0[8 + i] = h0[i];
          kb1[i] = l1[i]; kb1[8 + i] = h1[i];
        }
      }
      v8f cc = {};
      cc = __builtin_amdgcn_wmma_f32_16x16x32_bf16(false, qa0, false, kb0, (short)0, cc, false, false);
      cc = __builtin_amdgcn_wmma_f32_16x16x32_bf16(false, qa1, false, kb1, (short)0, cc, false, false);
      int kk = kt * 16 + idx;
      int sk = seg[b * SS + kk];
#pragma unroll
      for (int r = 0; r < 8; ++r) {
        int qq = qbase + r + 8 * hi;
        bool ok = (kk <= qq) && (sk == segq[r]);
        float v = ok ? cc[r] * scale : -1e30f;
        if (sub == 0) s0[r] = v; else s1[r] = v;
      }
    }

#pragma unroll
    for (int r = 0; r < 8; ++r) {
      float mx = s0[r] > s1[r] ? s0[r] : s1[r];
#pragma unroll
      for (int m = 1; m < 16; m <<= 1) {
        float o = __shfl_xor(mx, m, 32);
        mx = mx > o ? mx : o;
      }
      float mnew = mold[r] > mx ? mold[r] : mx;
      float f = __expf(mold[r] - mnew);
      mold[r] = mnew;
      float p0 = __expf(s0[r] - mnew);
      float p1 = __expf(s1[r] - mnew);
      float rs = p0 + p1;
#pragma unroll
      for (int m = 1; m < 16; m <<= 1) rs += __shfl_xor(rs, m, 32);
      lsum[r] = lsum[r] * f + rs;
#pragma unroll
      for (int t = 0; t < 4; ++t) acc[t][r] *= f;
      P[wave][r + 8 * hi][idx] = (bf16)p0;
      P[wave][r + 8 * hi][16 + idx] = (bf16)p1;
    }
    LDS_FENCE();

    v16bf pa;
    {
      v8bf lo = *(const v8bf*)(&P[wave][idx][hi * 8]);
      v8bf hs = *(const v8bf*)(&P[wave][idx][16 + hi * 8]);
#pragma unroll
      for (int i = 0; i < 8; ++i) { pa[i] = lo[i]; pa[8 + i] = hs[i]; }
    }

    int kbase = p * 32;
#pragma unroll
    for (int t = 0; t < 4; ++t) {
      const bf16* vrow = vbase + (size_t)(t * 16 + idx) * SS + kbase;
      v8bf lo = *(const v8bf*)(vrow + hi * 8);
      v8bf hs = *(const v8bf*)(vrow + 16 + hi * 8);
      v16bf vb;
#pragma unroll
      for (int i = 0; i < 8; ++i) { vb[i] = lo[i]; vb[8 + i] = hs[i]; }
      acc[t] = __builtin_amdgcn_wmma_f32_16x16x32_bf16(false, pa, false, vb,
                                                       (short)0, acc[t], false, false);
    }
    LDS_FENCE();
  }

#pragma unroll
  for (int t = 0; t < 4; ++t) {
#pragma unroll
    for (int r = 0; r < 8; ++r) {
      int qq = qbase + r + 8 * hi;
      float v = acc[t][r] / lsum[r];
      hoB[((size_t)(b * SS + qq)) * EE + h * HDD + t * 16 + idx] = (bf16)v;
    }
  }
}

// ---------------------------------------------------------------------------
// x = LayerNorm(in)*g + b + resid ; also bf16 copy. One block per row (E=512).
// ---------------------------------------------------------------------------
__global__ __launch_bounds__(256) void k_ln_res(const float* __restrict__ in,
                                                const float* __restrict__ resid,
                                                const float* __restrict__ g,
                                                const float* __restrict__ be,
                                                float* __restrict__ xout,
                                                bf16* __restrict__ xbf) {
  __shared__ float red[256];
  int row = blockIdx.x, t = threadIdx.x;
  const float* rp = in + (size_t)row * EE;
  float v0 = rp[t], v1 = rp[t + 256];
  red[t] = v0 + v1;
  __syncthreads();
  for (int o = 128; o > 0; o >>= 1) {
    if (t < o) red[t] += red[t + o];
    __syncthreads();
  }
  float mean = red[0] * (1.0f / EE);
  __syncthreads();
  float d0 = v0 - mean, d1 = v1 - mean;
  red[t] = d0 * d0 + d1 * d1;
  __syncthreads();
  for (int o = 128; o > 0; o >>= 1) {
    if (t < o) red[t] += red[t + o];
    __syncthreads();
  }
  float inv = rsqrtf(red[0] * (1.0f / EE) + 1e-6f);
#pragma unroll
  for (int i = 0; i < 2; ++i) {
    int e = t + i * 256;
    float d = (i == 0) ? d0 : d1;
    float v = d * inv * g[e] + be[e] + resid[(size_t)row * EE + e];
    xout[(size_t)row * EE + e] = v;
    xbf[(size_t)row * EE + e] = (bf16)v;
  }
}

// ---------------------------------------------------------------------------
// Host orchestration
// ---------------------------------------------------------------------------
extern "C" void kernel_launch(void* const* d_in, const int* in_sizes, int n_in,
                              void* d_out, int out_size, void* d_ws, size_t ws_size,
                              hipStream_t stream) {
  const int* tokens = (const int*)d_in[0];
  const float* word_embed = (const float*)d_in[1];
  const float* pos_embed = (const float*)d_in[2];
  const float* KQV = (const float*)d_in[3];
  const float* WO = (const float*)d_in[4];
  const float* W_up = (const float*)d_in[5];
  const float* b_up = (const float*)d_in[6];
  const float* W_down = (const float*)d_in[7];
  const float* b_down = (const float*)d_in[8];
  const float* g1 = (const float*)d_in[9];
  const float* be1 = (const float*)d_in[10];
  const float* g2 = (const float*)d_in[11];
  const float* be2 = (const float*)d_in[12];
  const float* unembed_b = (const float*)d_in[13];
  const int* start_id = (const int*)d_in[14];

  const int M = BB * SS;  // 4096
  char* base = (char*)d_ws;
  size_t off = 0;
  auto alloc = [&](size_t bytes) -> char* {
    char* p = base + off;
    off = (off + bytes + 255) & ~(size_t)255;
    return p;
  };

  int* relpos   = (int*)alloc((size_t)M * 4);
  int* seg      = (int*)alloc((size_t)M * 4);
  float* x_f32  = (float*)alloc((size_t)M * EE * 4);
  bf16* x_bf    = (bf16*)alloc((size_t)M * EE * 2);
  bf16* kqv_bf  = (bf16*)alloc((size_t)M * T3 * 2);
  bf16* vT      = (bf16*)alloc((size_t)BB * HH * HDD * SS * 2);
  bf16* ho_bf   = (bf16*)alloc((size_t)M * EE * 2);
  float* proj   = (float*)alloc((size_t)M * EE * 4);
  bf16* h_bf    = (bf16*)alloc((size_t)M * FFN * 2);
  float* y_f32  = (float*)alloc((size_t)M * EE * 4);
  bf16* kqvT    = (bf16*)alloc((size_t)LLY * T3 * EE * 2);
  bf16* woT     = (bf16*)alloc((size_t)LLY * EE * EE * 2);
  bf16* wupT    = (bf16*)alloc((size_t)LLY * FFN * EE * 2);
  bf16* wdownT  = (bf16*)alloc((size_t)LLY * EE * FFN * 2);
  bf16* wemb_bf = (bf16*)alloc((size_t)VV * EE * 2);

  k_scan<<<BB, 32, 0, stream>>>(tokens, start_id, relpos, seg);
  k_embed<<<M, 256, 0, stream>>>(tokens, relpos, word_embed, pos_embed, x_f32, x_bf);

  for (int l = 0; l < LLY; ++l) {
    size_t n;
    n = (size_t)EE * T3;
    k_transpose_bf<<<(int)((n + 255) / 256), 256, 0, stream>>>(
        KQV + (size_t)l * n, kqvT + (size_t)l * n, EE, T3);
    n = (size_t)EE * EE;
    k_transpose_bf<<<(int)((n + 255) / 256), 256, 0, stream>>>(
        WO + (size_t)l * n, woT + (size_t)l * n, EE, EE);
    n = (size_t)EE * FFN;
    k_transpose_bf<<<(int)((n + 255) / 256), 256, 0, stream>>>(
        W_up + (size_t)l * n, wupT + (size_t)l * n, EE, FFN);
    n = (size_t)FFN * EE;
    k_transpose_bf<<<(int)((n + 255) / 256), 256, 0, stream>>>(
        W_down + (size_t)l * n, wdownT + (size_t)l * n, FFN, EE);
  }
  {
    size_t n = (size_t)VV * EE;
    k_f2bf<<<(int)((n + 255) / 256), 256, 0, stream>>>(word_embed, wemb_bf, n);
  }

  for (int l = 0; l < LLY; ++l) {
    k_gemm_bt<<<dim3(T3 / 256, M / 128), 256, 0, stream>>>(
        x_bf, kqvT + (size_t)l * T3 * EE, nullptr, nullptr, kqv_bf, M, T3, EE, 0);
    {
      size_t n = (size_t)BB * HH * HDD * SS;
      k_packVT<<<(int)((n + 255) / 256), 256, 0, stream>>>(kqv_bf, vT);
    }
    k_attn<<<BB * HH * (SS / 64), 128, 0, stream>>>(kqv_bf, vT, seg, ho_bf);
    k_gemm_bt<<<dim3(EE / 256, M / 128), 256, 0, stream>>>(
        ho_bf, woT + (size_t)l * EE * EE, nullptr, proj, nullptr, M, EE, EE, 0);
    k_ln_res<<<M, 256, 0, stream>>>(proj, x_f32, g1 + (size_t)l * EE,
                                    be1 + (size_t)l * EE, x_f32, x_bf);
    k_gemm_bt<<<dim3(FFN / 256, M / 128), 256, 0, stream>>>(
        x_bf, wupT + (size_t)l * FFN * EE, b_up + (size_t)l * FFN,
        nullptr, h_bf, M, FFN, EE, 1);
    k_gemm_bt<<<dim3(EE / 256, M / 128), 256, 0, stream>>>(
        h_bf, wdownT + (size_t)l * EE * FFN, b_down + (size_t)l * EE,
        y_f32, nullptr, M, EE, FFN, 0);
    k_ln_res<<<M, 256, 0, stream>>>(y_f32, x_f32, g2 + (size_t)l * EE,
                                    be2 + (size_t)l * EE, x_f32, x_bf);
  }

  k_gemm_bt<<<dim3(VV / 256, M / 128), 256, 0, stream>>>(
      x_bf, wemb_bf, unembed_b, (float*)d_out, nullptr, M, VV, EE, 0);
}
